// PointFinbaEncoder_42915313222100
// MI455X (gfx1250) — matrix operand order repaired and checked by reading, not compiled, not run on hardware
//
#include <hip/hip_runtime.h>

typedef __bf16 bf16t;
typedef __attribute__((ext_vector_type(16))) __bf16 v16bf;
typedef __attribute__((ext_vector_type(8)))  float  v8f;

#define DEV static __device__ __forceinline__

static constexpr int ROWS = 16 * 512;   // B*N tokens
static constexpr int C    = 384;
static constexpr int T    = 512;
static constexpr int H    = 6, HD = 64;
static constexpr int DIN  = 768, DST = 16, DTR = 24;
static constexpr int TML  = 32;
static constexpr int XPN  = 56;         // x_proj output width
static constexpr int KRMAX = 1344;      // largest K over all GEMMs

enum { ACT_NONE = 0, ACT_SILU, ACT_TANH, ACT_SIG, ACT_RELU2, ACT_SOFTPLUS, ACT_EXPNEGEXP };

DEV float siluf(float x) { return x * __builtin_amdgcn_rcpf(1.0f + __expf(-x)); }
DEV float sigf(float x)  { return __builtin_amdgcn_rcpf(1.0f + __expf(-x)); }

DEV bf16t f2bf(float f) {
  unsigned u = __builtin_bit_cast(unsigned, f);
  unsigned short h = (unsigned short)((u + 0x7fffu + ((u >> 16) & 1u)) >> 16);
  return __builtin_bit_cast(bf16t, h);
}

DEV float wred(float x) {
#pragma unroll
  for (int m = 16; m; m >>= 1) x += __shfl_xor(x, m, 32);
  return x;
}

// ---------------- elementwise ----------------
__global__ void k_copy(float* o, const float* a, int n) {
  int i = blockIdx.x * 256 + threadIdx.x; if (i < n) o[i] = a[i];
}
__global__ void k_add(float* o, const float* a, const float* b, int n) {
  int i = blockIdx.x * 256 + threadIdx.x; if (i < n) o[i] = a[i] + b[i];
}
__global__ void k_mul(float* o, const float* a, const float* b, int n) {
  int i = blockIdx.x * 256 + threadIdx.x; if (i < n) o[i] = a[i] * b[i];
}
__global__ void k_fma3(float* o, const float* base, const float* a, const float* b, int n) {
  int i = blockIdx.x * 256 + threadIdx.x; if (i < n) o[i] = base[i] + a[i] * b[i];
}
// xx = shift(x) - x  (shift is causal within each length-512 sequence)
__global__ void k_shift(float* xx, const float* src, int n) {
  int i = blockIdx.x * 256 + threadIdx.x; if (i >= n) return;
  int row = i / C; int t = row & (T - 1);
  xx[i] = (t ? src[i - C] : 0.0f) - src[i];
}
// out = base + xx * (maa[c] + mm[i])
__global__ void k_mixvm(float* o, const float* base, const float* xx,
                        const float* maa, const float* mm, int n) {
  int i = blockIdx.x * 256 + threadIdx.x; if (i >= n) return;
  int c = i % C;
  float m = maa[c];
  if (mm) m += mm[i];
  o[i] = base[i] + xx[i] * m;
}
__global__ void k_glu(const float* fc1, float* o, int n) {
  int i = blockIdx.x * 256 + threadIdx.x; if (i >= n) return;
  int row = i / 1024, j = i % 1024;
  float a = fc1[(size_t)row * 2048 + j];
  float g = fc1[(size_t)row * 2048 + 1024 + j];
  o[i] = a * siluf(g);
}

// ---------------- layernorm / groupnorm ----------------
template <int CD>
__global__ void k_ln(const float* __restrict__ in, const float* __restrict__ w,
                     const float* __restrict__ b, float* __restrict__ out,
                     int R, float eps) {
  int row = blockIdx.x * (blockDim.x >> 5) + (threadIdx.x >> 5);
  if (row >= R) return;
  const int lane = threadIdx.x & 31;
  constexpr int NV = CD / 32;
  float v[NV];
  const float* p = in + (size_t)row * CD;
#pragma unroll
  for (int i = 0; i < NV; ++i) v[i] = p[lane + i * 32];
  float s = 0;
#pragma unroll
  for (int i = 0; i < NV; ++i) s += v[i];
  s = wred(s);
  const float mean = s / CD;
  float q = 0;
#pragma unroll
  for (int i = 0; i < NV; ++i) { float d = v[i] - mean; q += d * d; }
  q = wred(q);
  const float rstd = rsqrtf(q / CD + eps);
  float* o = out + (size_t)row * CD;
#pragma unroll
  for (int i = 0; i < NV; ++i) {
    int c = lane + i * 32;
    o[c] = (v[i] - mean) * rstd * w[c] + b[c];
  }
}

__global__ void k_gn(const float* __restrict__ in, const float* __restrict__ w,
                     const float* __restrict__ b, float* __restrict__ out, int R) {
  int row = blockIdx.x * 8 + (threadIdx.x >> 5);
  if (row >= R) return;
  int lane = threadIdx.x & 31;
  const float* p = in + (size_t)row * HD;
  float v0 = p[lane], v1 = p[lane + 32];
  float mean = wred(v0 + v1) * (1.0f / HD);
  float d0 = v0 - mean, d1 = v1 - mean;
  float q = wred(d0 * d0 + d1 * d1);
  float rstd = rsqrtf(q * (1.0f / HD) + 0.00064f);
  int h = row % H;
  int c0 = h * HD + lane, c1 = c0 + 32;
  float* o = out + (size_t)row * HD;
  o[lane]      = d0 * rstd * w[c0] + b[c0];
  o[lane + 32] = d1 * rstd * w[c1] + b[c1];
}

// ---------------- KNN (top-8 nearest, ties keep lower index) ----------------
__global__ void k_knn(const float* __restrict__ center, int* __restrict__ idxf,
                      float* __restrict__ dist) {
  int g = blockIdx.x * 256 + threadIdx.x;
  if (g >= ROWS) return;
  int b = g >> 9;
  const float cx = center[g * 3 + 0], cy = center[g * 3 + 1], cz = center[g * 3 + 2];
  float bd[8]; int bi[8];
#pragma unroll
  for (int i = 0; i < 8; ++i) { bd[i] = 3.4e38f; bi[i] = 0; }
  const float* cb = center + (size_t)b * 512 * 3;
  for (int j = 0; j < 512; ++j) {
    float dx = cx - cb[j * 3 + 0], dy = cy - cb[j * 3 + 1], dz = cz - cb[j * 3 + 2];
    float d2 = dx * dx + dy * dy + dz * dz;
    if (d2 < bd[7]) {
      int p = 7;
      while (p > 0 && d2 < bd[p - 1]) { bd[p] = bd[p - 1]; bi[p] = bi[p - 1]; --p; }
      bd[p] = d2; bi[p] = j;
    }
  }
#pragma unroll
  for (int i = 0; i < 8; ++i) {
    idxf[g * 8 + i] = b * 512 + bi[i];
    dist[g * 8 + i] = sqrtf(fmaxf(bd[i], 0.0f));
  }
}

// ---------------- GLG: deterministic global std + fused gather/normalize/max ----------------
__global__ void k_glgstats(const float* __restrict__ feat, const int* __restrict__ idxf,
                           float* __restrict__ pS, float* __restrict__ pQ) {
  const long long TOT = (long long)ROWS * 8 * C;
  float s = 0, q = 0;
  for (long long e = (long long)blockIdx.x * 256 + threadIdx.x; e < TOT;
       e += (long long)gridDim.x * 256) {
    int c = (int)(e % C);
    long long t = e / C;
    int k = (int)(t % 8);
    int row = (int)(t / 8);
    float f = feat[(size_t)row * C + c];
    float kx = feat[(size_t)idxf[row * 8 + k] * C + c];
    float d = kx - f;
    s += d; q += d * d;
  }
  __shared__ float ss[256], qq[256];
  ss[threadIdx.x] = s; qq[threadIdx.x] = q;
  __syncthreads();
  for (int st = 128; st; st >>= 1) {
    if (threadIdx.x < st) { ss[threadIdx.x] += ss[threadIdx.x + st]; qq[threadIdx.x] += qq[threadIdx.x + st]; }
    __syncthreads();
  }
  if (!threadIdx.x) { pS[blockIdx.x] = ss[0]; pQ[blockIdx.x] = qq[0]; }
}

__global__ void k_statreduce(const float* pS, const float* pQ, float* stats) {
  __shared__ float ss[256], qq[256];
  float s = 0, q = 0;
  for (int i = threadIdx.x; i < 1024; i += 256) { s += pS[i]; q += pQ[i]; }
  ss[threadIdx.x] = s; qq[threadIdx.x] = q;
  __syncthreads();
  for (int st = 128; st; st >>= 1) {
    if (threadIdx.x < st) { ss[threadIdx.x] += ss[threadIdx.x + st]; qq[threadIdx.x] += qq[threadIdx.x + st]; }
    __syncthreads();
  }
  if (!threadIdx.x) {
    const double M = (double)ROWS * 8 * C;
    double var = ((double)qq[0] - (double)ss[0] * (double)ss[0] / M) / (M - 1.0);
    float stdv = (float)sqrt(var > 0 ? var : 0);
    stats[0] = 1.0f / (stdv + 1e-5f);
  }
}

__global__ void k_glgapply(const float* __restrict__ feat, const int* __restrict__ idxf,
                           const float* __restrict__ dist, const float* __restrict__ alpha,
                           const float* __restrict__ beta, const float* __restrict__ stats,
                           float* __restrict__ out, int n) {
  int i = blockIdx.x * 256 + threadIdx.x; if (i >= n) return;
  int row = i / C, c = i % C;
  float f = feat[i];
  float inv = stats[0];
  float m1 = -3.4e38f, m2 = -3.4e38f;
#pragma unroll
  for (int k = 0; k < 8; ++k) {
    int j = idxf[row * 8 + k];
    float kx = feat[(size_t)j * C + c];
    float dd = dist[row * 8 + k];
    float wgt = __expf(-0.5f * dd * dd);
    float v1 = (alpha[c] * ((kx - f) * inv) + beta[c]) * wgt;
    float v2 = (alpha[C + c] * f + beta[C + c]) * wgt;
    m1 = fmaxf(m1, v1); m2 = fmaxf(m2, v2);
  }
  out[(size_t)row * 2 * C + c]     = m1;
  out[(size_t)row * 2 * C + C + c] = m2;
}

// ---------------- bf16 staging ----------------
// Weights: Wt[k*Nr+n] = W[n,k] (trans=0, stored [N,K]) or W[k,n] (trans=1, stored [K,N]);
// zero-padded in K (to Kr) and N (to Nr) so the GEMM needs no K/N guards on loads.
__global__ void k_wconv(const float* __restrict__ W, bf16t* __restrict__ Wt,
                        int N, int K, int Nr, int tot, int trans) {
  int i = blockIdx.x * 256 + threadIdx.x; if (i >= tot) return;
  int k = i / Nr, n = i % Nr;
  float v = 0.0f;
  if (k < K && n < N) v = trans ? W[(size_t)k * N + n] : W[(size_t)n * K + k];
  Wt[i] = f2bf(v);
}
// Activations: Ab[m*Kr+k] = A[m*lda+k], zero-padded K tail. Converted once per GEMM
// instead of once per (tile-column x chunk) inside the GEMM loop.
__global__ void k_aconv(const float* __restrict__ A, int lda, bf16t* __restrict__ Ab,
                        int K, int Kr, long long tot) {
  long long i = (long long)blockIdx.x * 256 + threadIdx.x; if (i >= tot) return;
  int k = (int)(i % Kr);
  long long m = i / Kr;
  float v = (k < K) ? A[m * lda + k] : 0.0f;
  Ab[i] = f2bf(v);
}

// ---------------- WMMA GEMM: Out[M=8192,N] = act(A * W^T + bias) (+resid), bf16 operands ----
// Each wave computes a 16x32 output strip: one A fragment feeds two v_wmma issues.
// All 4 waves of a block share the same n0 (tile count is a multiple of 4 and 512 % 4 == 0),
// so the 32x32 bf16 B chunk is staged once per block through a double-buffered LDS tile
// (one s_barrier per K chunk) and read back one bf16 at a time (stride 32 along k).
template <int ACT>
__global__ void __launch_bounds__(128)
k_gemm(const bf16t* __restrict__ Ab,
       const bf16t* __restrict__ Wt, int Nr,
       const float* __restrict__ bias, const float* __restrict__ resid,
       float* __restrict__ Out, int N, int Kr) {
  __shared__ __align__(32) bf16t Bs[2][32 * 32];
  const int wave = threadIdx.x >> 5;
  const int lane = threadIdx.x & 31;
  const int mt = ROWS >> 4;
  const int ntp = (N + 31) >> 5;                  // 32-wide tile pairs
  const int tile = blockIdx.x * 4 + wave;
  if (tile >= mt * ntp) return;                   // never taken (full blocks)
  const int m0 = (tile % mt) << 4;
  const int n0 = (tile / mt) << 5;                // identical across the block
  const int half = lane >> 4;
  const int l15 = lane & 15;
  const bf16t* arow = Ab + (size_t)(m0 + l15) * Kr;
  const int srow = threadIdx.x >> 2;              // cooperative B staging: 32 rows
  const int scol = (threadIdx.x & 3) * 8;         // x 4 groups of 8 bf16 (16B)

  v8f acc0 = {}, acc1 = {};
  for (int kc = 0; kc < Kr; kc += 32) {
    const int pp = (kc >> 5) & 1;
    const uint4 wv = *(const uint4*)(Wt + (size_t)(kc + srow) * Nr + n0 + scol);
    *(uint4*)(&Bs[pp][srow * 32 + scol]) = wv;

    const int b1 = kc + half * 8;  // lane covers K in [b1,b1+8) and [b1+16,b1+24)
    union { v16bf v; uint4 q[2]; } af;
    af.q[0] = *(const uint4*)(arow + b1);
    af.q[1] = *(const uint4*)(arow + b1 + 16);
    __builtin_prefetch(arow + kc + 32, 0, 0);     // global_prefetch_b8 next K chunk
    __syncthreads();                              // B chunk visible; prev buffer free

    v16bf b0, b1v;
#pragma unroll
    for (int s = 0; s < 16; ++s) {
      const int kk = half * 8 + (s & 7) + (s >> 3) * 16;
      b0[s]  = Bs[pp][kk * 32 + l15];
      b1v[s] = Bs[pp][kk * 32 + 16 + l15];
    }
    acc0 = __builtin_amdgcn_wmma_f32_16x16x32_bf16(false, af.v, false, b0,
                                                   (short)0, acc0, false, false);
    acc1 = __builtin_amdgcn_wmma_f32_16x16x32_bf16(false, af.v, false, b1v,
                                                   (short)0, acc1, false, false);
  }

#pragma unroll 2
  for (int tnum = 0; tnum < 2; ++tnum) {
    const int n = n0 + tnum * 16 + l15;
    if (n >= N) continue;
    const v8f acc = tnum ? acc1 : acc0;
    const float bv = bias ? bias[n] : 0.0f;
#pragma unroll
    for (int r = 0; r < 8; ++r) {
      float v = acc[r] + bv;
      if constexpr (ACT == ACT_SILU)           v = siluf(v);
      else if constexpr (ACT == ACT_TANH)      v = tanhf(v);
      else if constexpr (ACT == ACT_SIG)       v = sigf(v);
      else if constexpr (ACT == ACT_RELU2)     { float t = fmaxf(v, 0.0f); v = t * t; }
      else if constexpr (ACT == ACT_SOFTPLUS)  v = (v > 20.0f) ? v : logf(1.0f + __expf(v));
      else if constexpr (ACT == ACT_EXPNEGEXP) v = __expf(-__expf(v));
      const size_t o = (size_t)(m0 + r + half * 8) * N + n;
      if (resid) v += resid[o];
      Out[o] = v;
    }
  }
}

// ---------------- RWKV wkv scan: one block per (b,h), thread j owns state column s[:,j] ----------------
__global__ void __launch_bounds__(64)
k_wkv(const float* __restrict__ r, const float* __restrict__ k,
      const float* __restrict__ v, const float* __restrict__ w,
      const float* __restrict__ u, float* __restrict__ y) {
  const int b = blockIdx.x / H, h = blockIdx.x % H;
  const int j = threadIdx.x;
  __shared__ float rr[HD], kk[HD], ww[HD], uu[HD];
  uu[j] = u[h * HD + j];
  __syncthreads();
  float s[HD];
#pragma unroll
  for (int i = 0; i < HD; ++i) s[i] = 0.0f;
  for (int t = 0; t < T; ++t) {
    const size_t base = (size_t)(b * T + t) * C + h * HD;
    rr[j] = r[base + j]; kk[j] = k[base + j]; ww[j] = w[base + j];
    const float vj = v[base + j];
    __syncthreads();
    float acc = 0.0f;
#pragma unroll
    for (int i = 0; i < HD; ++i) {
      float kv = kk[i] * vj;
      acc += rr[i] * (uu[i] * kv + s[i]);
      s[i] = ww[i] * s[i] + kv;
    }
    y[base + j] = acc;
    __syncthreads();
  }
}

// ---------------- Mamba selective scan: thread per channel d, 16-state register file ----------------
__global__ void __launch_bounds__(256)
k_mamba(const float* __restrict__ xm, const float* __restrict__ dt,
        const float* __restrict__ xdb, const float* __restrict__ Alog,
        const float* __restrict__ Dv, const float* __restrict__ xz,
        float* __restrict__ out) {
  const int b = blockIdx.x / 3;
  const int d = (blockIdx.x % 3) * 256 + threadIdx.x;
  __shared__ float Bs[DST], Cs[DST];
  float a[DST], h[DST];
#pragma unroll
  for (int s = 0; s < DST; ++s) { a[s] = -__expf(Alog[d * DST + s]); h[s] = 0.0f; }
  const float dval = Dv[d];
  for (int t = 0; t < T; ++t) {
    const int row = b * T + t;
    if (threadIdx.x < DST) Bs[threadIdx.x] = xdb[(size_t)row * XPN + DTR + threadIdx.x];
    else if (threadIdx.x < 2 * DST)
      Cs[threadIdx.x - DST] = xdb[(size_t)row * XPN + DTR + DST + (threadIdx.x - DST)];
    __syncthreads();
    const float dtv = dt[(size_t)row * DIN + d];
    const float xv  = xm[(size_t)row * DIN + d];
    float yy = 0.0f;
#pragma unroll
    for (int s = 0; s < DST; ++s) {
      h[s] = __expf(dtv * a[s]) * h[s] + dtv * Bs[s] * xv;
      yy += h[s] * Cs[s];
    }
    const float zv = xz[(size_t)row * 1536 + DIN + d];
    out[(size_t)row * DIN + d] = (yy + xv * dval) * siluf(zv);
    __syncthreads();
  }
}

// depthwise causal conv(4) + bias + silu; input = xz[:, :768] with row stride 1536
__global__ void k_conv(const float* __restrict__ xz, const float* __restrict__ cw,
                       const float* __restrict__ cb, float* __restrict__ out, int n) {
  int i = blockIdx.x * 256 + threadIdx.x; if (i >= n) return;
  const int d = i % DIN, row = i / DIN;
  const int t = row & (T - 1);
  float acc = cb[d];
#pragma unroll
  for (int j = 0; j < 4; ++j) {
    int tt = t - 3 + j;
    if (tt >= 0) acc += cw[d * 4 + j] * xz[(size_t)(row - 3 + j) * 1536 + d];
  }
  out[i] = siluf(acc);
}

// ---------------- host side ----------------
static void gemm(hipStream_t st, bf16t* wbuf, bf16t* abuf, const float* A, int lda,
                 const float* Wsrc, int trans, int N, int K,
                 const float* bias, int act, const float* resid, float* Out) {
  const int Nr = (N + 31) & ~31;
  const int Kr = (K + 31) & ~31;
  const int tot = Nr * Kr;
  k_wconv<<<(tot + 255) / 256, 256, 0, st>>>(Wsrc, wbuf, N, K, Nr, tot, trans);
  const long long atot = (long long)ROWS * Kr;
  k_aconv<<<(unsigned)((atot + 255) / 256), 256, 0, st>>>(A, lda, abuf, K, Kr, atot);
  const int tiles = (ROWS / 16) * ((N + 31) / 32);
  dim3 g((tiles + 3) / 4), blk(128);
  switch (act) {
    case ACT_SILU:      k_gemm<ACT_SILU>     <<<g, blk, 0, st>>>(abuf, wbuf, Nr, bias, resid, Out, N, Kr); break;
    case ACT_TANH:      k_gemm<ACT_TANH>     <<<g, blk, 0, st>>>(abuf, wbuf, Nr, bias, resid, Out, N, Kr); break;
    case ACT_SIG:       k_gemm<ACT_SIG>      <<<g, blk, 0, st>>>(abuf, wbuf, Nr, bias, resid, Out, N, Kr); break;
    case ACT_RELU2:     k_gemm<ACT_RELU2>    <<<g, blk, 0, st>>>(abuf, wbuf, Nr, bias, resid, Out, N, Kr); break;
    case ACT_SOFTPLUS:  k_gemm<ACT_SOFTPLUS> <<<g, blk, 0, st>>>(abuf, wbuf, Nr, bias, resid, Out, N, Kr); break;
    case ACT_EXPNEGEXP: k_gemm<ACT_EXPNEGEXP><<<g, blk, 0, st>>>(abuf, wbuf, Nr, bias, resid, Out, N, Kr); break;
    default:            k_gemm<ACT_NONE>     <<<g, blk, 0, st>>>(abuf, wbuf, Nr, bias, resid, Out, N, Kr); break;
  }
}

#define EW(kern, n, ...) kern<<<((n) + 255) / 256, 256, 0, stream>>>(__VA_ARGS__, (n))

extern "C" void kernel_launch(void* const* d_in, const int* in_sizes, int n_in,
                              void* d_out, int out_size, void* d_ws, size_t ws_size,
                              hipStream_t stream) {
  (void)in_sizes; (void)n_in; (void)out_size; (void)ws_size;
  auto f = [&](int i) { return (const float*)d_in[i]; };

  // workspace carve (bump allocator, 256B aligned)
  char* wp = (char*)d_ws;
  auto carve = [&](size_t bytes) -> void* {
    void* r = (void*)wp; wp += (bytes + 255) & ~(size_t)255; return r;
  };
  float* S[10];
  for (int i = 0; i < 10; ++i) S[i] = (float*)carve((size_t)ROWS * C * 4);
  float* d1   = (float*)carve((size_t)ROWS * DIN * 4);   // glg 2C out / mamba xm
  float* d2   = (float*)carve((size_t)ROWS * DIN * 4);   // glg ln out / mamba dt
  float* big1 = (float*)carve((size_t)ROWS * 2048 * 4);  // fc1 / xz / cmix kk
  float* big2 = (float*)carve((size_t)ROWS * 1024 * 4);  // glu / mamba y
  float* xb   = (float*)carve((size_t)ROWS * C * 4);     // x + pos (residual base)
  float* lora = (float*)carve((size_t)ROWS * 160 * 4);
  float* td64 = (float*)carve((size_t)ROWS * 64 * 4);
  float* xdbv = (float*)carve((size_t)ROWS * XPN * 4);
  float* kdist = (float*)carve((size_t)ROWS * 8 * 4);
  int*   kidx  = (int*)carve((size_t)ROWS * 8 * 4);
  float* pS = (float*)carve(1024 * 4);
  float* pQ = (float*)carve(1024 * 4);
  float* stats = (float*)carve(256);
  bf16t* warena = (bf16t*)carve(4u * 1024 * 1024);             // bf16 weight slot (reused)
  bf16t* aarena = (bf16t*)carve((size_t)ROWS * KRMAX * 2);     // bf16 activation slot (reused)

  float* xcur = (float*)d_out;                            // residual stream lives in d_out
  const float* pos = f(2);
  const int nC = ROWS * C;

  EW(k_copy, nC, xcur, f(1));
  k_knn<<<(ROWS + 255) / 256, 256, 0, stream>>>(f(0), kidx, kdist);

  int p = 3;
  for (int blk = 0; blk < 4; ++blk) {
    const float *galpha = f(p), *gbeta = f(p + 1), *glnb = f(p + 2), *glnw = f(p + 3),
                *gmlpb = f(p + 4), *gmlpw = f(p + 5); p += 6;
    const float *ln0b = nullptr, *ln0w = nullptr;
    if (blk == 0) { ln0b = f(p); ln0w = f(p + 1); p += 2; }
    const float *ln1b = f(p), *ln1w = f(p + 1), *ln2b = f(p + 2), *ln2w = f(p + 3),
                *ln3b = f(p + 4), *ln3w = f(p + 5); p += 6;

    // ---- x + pos, optional ln0, GLG branch ----
    EW(k_add, nC, xb, xcur, pos);
    if (blk == 0) k_ln<384><<<(ROWS + 7) / 8, 256, 0, stream>>>(xb, ln0w, ln0b, xb, ROWS, 1e-5f);
    k_ln<384><<<(ROWS + 7) / 8, 256, 0, stream>>>(xb, ln1w, ln1b, S[0], ROWS, 1e-5f);
    k_glgstats<<<1024, 256, 0, stream>>>(S[0], kidx, pS, pQ);
    k_statreduce<<<1, 256, 0, stream>>>(pS, pQ, stats);
    EW(k_glgapply, nC, S[0], kidx, kdist, galpha, gbeta, stats, d1);
    k_ln<768><<<(ROWS + 7) / 8, 256, 0, stream>>>(d1, glnw, glnb, d2, ROWS, 1e-5f);
    gemm(stream, warena, aarena, d2, 768, gmlpw, 0, C, 768, gmlpb, ACT_SILU, xb, xcur);

    if (blk % 2 == 0) {
      const float *Wg = f(p), *Wk = f(p + 1), *Wo = f(p + 2), *Wr = f(p + 3), *Wv = f(p + 4),
                  *gnb = f(p + 5), *gnw = f(p + 6), *tdw1 = f(p + 7), *tdw2 = f(p + 8),
                  *tdecay = f(p + 9), *faaaa = f(p + 10), *tmw1 = f(p + 11), *tmw2 = f(p + 12),
                  *maag = f(p + 13), *maak = f(p + 14), *maar = f(p + 15), *maav = f(p + 16),
                  *maaw = f(p + 17), *maax = f(p + 18); p += 19;
      const float *cWk = f(p), *cWr = f(p + 1), *cWv = f(p + 2),
                  *cmaak = f(p + 3), *cmaar = f(p + 4); p += 5;

      // ---- tmix ----
      k_ln<384><<<(ROWS + 7) / 8, 256, 0, stream>>>(xcur, ln2w, ln2b, S[0], ROWS, 1e-5f);
      EW(k_shift, nC, S[1], S[0]);                               // xx
      EW(k_mixvm, nC, S[7], S[0], S[1], maax, (const float*)nullptr);
      gemm(stream, warena, aarena, S[7], C, tmw1, 1, 160, C, nullptr, ACT_TANH, nullptr, lora);
      const float* maas[5] = {maaw, maak, maav, maar, maag};
      float* dsts[5] = {S[2], S[3], S[4], S[5], S[6]};           // xw,xk,xv,xr,xg
      for (int fi = 0; fi < 5; ++fi) {
        gemm(stream, warena, aarena, lora + fi * TML, 160, tmw2 + (size_t)fi * TML * C, 1,
             C, TML, nullptr, ACT_NONE, nullptr, S[9]);
        EW(k_mixvm, nC, dsts[fi], S[0], S[1], maas[fi], S[9]);
      }
      gemm(stream, warena, aarena, S[2], C, tdw1, 1, 64, C, nullptr, ACT_TANH, nullptr, td64);
      gemm(stream, warena, aarena, td64, 64, tdw2, 1, C, 64, tdecay, ACT_EXPNEGEXP, nullptr, S[7]); // w
      gemm(stream, warena, aarena, S[5], C, Wr, 0, C, C, nullptr, ACT_NONE, nullptr, S[8]);         // r
      gemm(stream, warena, aarena, S[3], C, Wk, 0, C, C, nullptr, ACT_NONE, nullptr, S[9]);         // k
      gemm(stream, warena, aarena, S[4], C, Wv, 0, C, C, nullptr, ACT_NONE, nullptr, S[5]);         // v
      gemm(stream, warena, aarena, S[6], C, Wg, 0, C, C, nullptr, ACT_SILU, nullptr, S[2]);         // g
      k_wkv<<<16 * H, 64, 0, stream>>>(S[8], S[9], S[5], S[7], faaaa, S[3]);
      k_gn<<<(ROWS * H + 7) / 8, 256, 0, stream>>>(S[3], gnw, gnb, S[4], ROWS * H);
      EW(k_mul, nC, S[4], S[4], S[2]);
      gemm(stream, warena, aarena, S[4], C, Wo, 0, C, C, nullptr, ACT_NONE, xcur, xcur);

      // ---- cmix ----
      k_ln<384><<<(ROWS + 7) / 8, 256, 0, stream>>>(xcur, ln3w, ln3b, S[0], ROWS, 1e-5f);
      EW(k_shift, nC, S[1], S[0]);
      EW(k_mixvm, nC, S[2], S[0], S[1], cmaak, (const float*)nullptr);
      EW(k_mixvm, nC, S[3], S[0], S[1], cmaar, (const float*)nullptr);
      gemm(stream, warena, aarena, S[2], C, cWk, 0, 1344, C, nullptr, ACT_RELU2, nullptr, big1);
      gemm(stream, warena, aarena, big1, 1344, cWv, 0, C, 1344, nullptr, ACT_NONE, nullptr, S[4]);
      gemm(stream, warena, aarena, S[3], C, cWr, 0, C, C, nullptr, ACT_SIG, nullptr, S[5]);
      EW(k_fma3, nC, xcur, xcur, S[5], S[4]);
    } else {
      const float *Alog = f(p), *Dvp = f(p + 1), *convb = f(p + 2), *convw = f(p + 3),
                  *dtb = f(p + 4), *dtp = f(p + 5), *inp = f(p + 6), *outp = f(p + 7),
                  *xp = f(p + 8); p += 9;
      const float *fc1 = f(p), *fc2 = f(p + 1); p += 2;

      // ---- mamba ----
      k_ln<384><<<(ROWS + 7) / 8, 256, 0, stream>>>(xcur, ln2w, ln2b, S[0], ROWS, 1e-5f);
      gemm(stream, warena, aarena, S[0], C, inp, 0, 1536, C, nullptr, ACT_NONE, nullptr, big1);  // xz
      EW(k_conv, ROWS * DIN, big1, convw, convb, d1);                                            // xm
      gemm(stream, warena, aarena, d1, DIN, xp, 0, XPN, DIN, nullptr, ACT_NONE, nullptr, xdbv);
      gemm(stream, warena, aarena, xdbv, XPN, dtp, 0, DIN, DTR, dtb, ACT_SOFTPLUS, nullptr, d2); // dt
      k_mamba<<<16 * 3, 256, 0, stream>>>(d1, d2, xdbv, Alog, Dvp, big1, big2);
      gemm(stream, warena, aarena, big2, DIN, outp, 0, C, DIN, nullptr, ACT_NONE, xcur, xcur);

      // ---- gmlp ----
      k_ln<384><<<(ROWS + 7) / 8, 256, 0, stream>>>(xcur, ln3w, ln3b, S[0], ROWS, 1e-5f);
      gemm(stream, warena, aarena, S[0], C, fc1, 0, 2048, C, nullptr, ACT_NONE, nullptr, big1);
      EW(k_glu, ROWS * 1024, big1, big2);
      gemm(stream, warena, aarena, big2, 1024, fc2, 0, C, 1024, nullptr, ACT_NONE, xcur, xcur);
    }
  }
}
